// FaultGAT_83786222010866
// MI455X (gfx1250) — matrix operand order repaired and checked by reading, not compiled, not run on hardware
//
#include <hip/hip_runtime.h>

typedef __bf16 bf16;
typedef __attribute__((ext_vector_type(16))) __bf16 v16bf;
typedef __attribute__((ext_vector_type(8)))  float  v8f;

#define NEG_SLOPE 0.2f
#define CDIV(a, b) (((a) + (b) - 1) / (b))

// ---------- monotonic float <-> uint key for atomicMax-based segment max ----------
__device__ __forceinline__ unsigned f2key(float f) {
    unsigned u = __float_as_uint(f);
    return (u & 0x80000000u) ? ~u : (u | 0x80000000u);
}
__device__ __forceinline__ float key2f(unsigned k) {
    unsigned u = (k & 0x80000000u) ? (k & 0x7FFFFFFFu) : ~k;
    return __uint_as_float(u);
}

// edge -> (src, dst), with implicit self-loops appended after the E0 real edges
__device__ __forceinline__ void edge_sd(const long long* __restrict__ ei, int E0,
                                        int e, int& s, int& d) {
    if (e < E0) { s = (int)ei[e]; d = (int)ei[(size_t)E0 + e]; }
    else        { s = d = e - E0; }
}

// ---------- 0) f32 -> bf16 conversion of x and W1 ----------
__global__ void k_convert(const float* __restrict__ x, const float* __restrict__ W1,
                          bf16* __restrict__ xbf, bf16* __restrict__ w1bf,
                          int nx, int nw) {
    int i = blockIdx.x * blockDim.x + threadIdx.x;
    if (i < nx) xbf[i]  = (bf16)x[i];
    if (i < nw) w1bf[i] = (bf16)W1[i];
}

// ---------- 1) h1 = x @ W1  (M x 128) = (M x 128)(128 x 128), bf16 WMMA, f32 acc ----------
// One wave per 16x16 C tile, K=128 as 4 x v_wmma_f32_16x16x32_bf16.
__global__ void k_gemm1_wmma(const bf16* __restrict__ A, const bf16* __restrict__ B,
                             float* __restrict__ C, int M) {
    int lane = threadIdx.x & 31;
    int wid  = threadIdx.x >> 5;
    int tile = blockIdx.x * (blockDim.x >> 5) + wid;
    int mTiles = CDIV(M, 16);
    if (tile >= mTiles * 8) return;              // wave-uniform exit (EXEC stays full)
    int mt = tile >> 3, nt = tile & 7;
    int m0 = mt << 4,  n0 = nt << 4;

    int l16  = lane & 15;
    // A k order used here (contiguous, same as B): lanes 0-15 hold K 0..15, lanes 16-31 hold K 16..31
    int arow = m0 + l16;  if (arow >= M) arow = M - 1;   // clamp, keep EXEC full
    int akb  = (lane < 16) ? 0 : 16;
    // B k order used here (contiguous, same as A): lanes 0-15 hold K 0..15, lanes 16-31 hold K 16..31
    int bcol = n0 + l16;
    int bkb  = (lane < 16) ? 0 : 16;

    v8f acc = {};
#pragma unroll
    for (int kt = 0; kt < 4; ++kt) {
        int k0 = kt << 5;
        const bf16* ap = A + (size_t)arow * 128 + k0 + akb;
        v16bf av, bv;
#pragma unroll
        for (int i = 0; i < 8; ++i) { av[i] = ap[i]; av[i + 8] = ap[8 + i]; }
#pragma unroll
        for (int i = 0; i < 16; ++i) bv[i] = B[(size_t)(k0 + bkb + i) * 128 + bcol];
        acc = __builtin_amdgcn_wmma_f32_16x16x32_bf16(false, av, false, bv,
                                                      (short)0, acc, false, false);
    }
    // C layout: lane l holds col n0+l16, rows m0 + r (+8 for upper half-wave)
    int crow = m0 + ((lane < 16) ? 0 : 8);
#pragma unroll
    for (int r = 0; r < 8; ++r)
        if (crow + r < M) C[(size_t)(crow + r) * 128 + bcol] = acc[r];
}

// ---------- 2) per-node attention coefficients, layer 1 ----------
__global__ void k_alpha1(const float* __restrict__ h1, const float* __restrict__ a_src,
                         const float* __restrict__ a_dst, float* __restrict__ as,
                         float* __restrict__ ad, int N) {
    int i = blockIdx.x * blockDim.x + threadIdx.x;   // (n, h)
    if (i >= N * 4) return;
    int n = i >> 2, h = i & 3;
    const float* row = h1 + (size_t)n * 128 + h * 32;
    float ss = 0.f, sd = 0.f;
#pragma unroll
    for (int f = 0; f < 32; ++f) {
        ss += row[f] * a_src[h * 32 + f];
        sd += row[f] * a_dst[h * 32 + f];
    }
    as[i] = ss; ad[i] = sd;
}

// ---------- 3) segment max over edges, layer 1 ----------
__global__ void k_max1(const long long* __restrict__ ei, int E0, int Etot,
                       const float* __restrict__ as, const float* __restrict__ ad,
                       unsigned* __restrict__ mkey) {
    int i = blockIdx.x * blockDim.x + threadIdx.x;   // (e, h)
    if (i >= Etot * 4) return;
    int e = i >> 2, h = i & 3;
    int s, d; edge_sd(ei, E0, e, s, d);
    float x  = as[s * 4 + h] + ad[d * 4 + h];
    float lg = x > 0.f ? x : NEG_SLOPE * x;
    atomicMax(&mkey[d * 4 + h], f2key(lg));
}

// ---------- 4) exp + segment sum (denominator), cache unnormalized e ----------
__global__ void k_exp1(const long long* __restrict__ ei, int E0, int Etot,
                       const float* __restrict__ as, const float* __restrict__ ad,
                       const unsigned* __restrict__ mkey, float* __restrict__ denom,
                       float* __restrict__ ee) {
    int i = blockIdx.x * blockDim.x + threadIdx.x;   // (e, h)
    if (i >= Etot * 4) return;
    int e = i >> 2, h = i & 3;
    int s, d; edge_sd(ei, E0, e, s, d);
    float x  = as[s * 4 + h] + ad[d * 4 + h];
    float lg = x > 0.f ? x : NEG_SLOPE * x;
    float ev = __expf(lg - key2f(mkey[d * 4 + h]));
    ee[i] = ev;
    atomicAdd(&denom[d * 4 + h], ev);
}

// ---------- 5) weighted scatter-add aggregation, layer 1. One wave32 per edge ----------
__global__ void k_agg1(const long long* __restrict__ ei, int E0, int Etot,
                       const float* __restrict__ h1, const float* __restrict__ ee,
                       const float* __restrict__ denom, float* __restrict__ out1) {
    int t = blockIdx.x * blockDim.x + threadIdx.x;
    int e = t >> 5;                                  // wave per edge
    int lane = t & 31;
    if (e >= Etot) return;
    int s, d; edge_sd(ei, E0, e, s, d);
#pragma unroll
    for (int j = 0; j < 4; ++j) {                    // j == head, c = j*32+lane
        int c = j * 32 + lane;
        float w = ee[(size_t)e * 4 + j] / denom[d * 4 + j];
        atomicAdd(&out1[(size_t)d * 128 + c], h1[(size_t)s * 128 + c] * w);
    }
}

// ---------- 6) bias + ELU (in place): x2 = elu(out1 + b1) ----------
__global__ void k_elu(float* __restrict__ x2, const float* __restrict__ b1, int n128) {
    int i = blockIdx.x * blockDim.x + threadIdx.x;
    if (i >= n128) return;
    float v = x2[i] + b1[i & 127];
    x2[i] = v > 0.f ? v : (__expf(v) - 1.f);
}

// ---------- 7) layer-2 projection (128 x 6) + attention coefficients ----------
__global__ void k_proj2(const float* __restrict__ x2, const float* __restrict__ W2,
                        const float* __restrict__ a2s, const float* __restrict__ a2d,
                        float* __restrict__ h2, float* __restrict__ as2,
                        float* __restrict__ ad2, int N) {
    int n = blockIdx.x * blockDim.x + threadIdx.x;
    if (n >= N) return;
    float acc[6] = {0.f, 0.f, 0.f, 0.f, 0.f, 0.f};
    const float* row = x2 + (size_t)n * 128;
    for (int k = 0; k < 128; ++k) {
        float xv = row[k];
#pragma unroll
        for (int c = 0; c < 6; ++c) acc[c] += xv * W2[k * 6 + c];
    }
    float ss = 0.f, sd = 0.f;
#pragma unroll
    for (int c = 0; c < 6; ++c) {
        h2[(size_t)n * 6 + c] = acc[c];
        ss += acc[c] * a2s[c];
        sd += acc[c] * a2d[c];
    }
    as2[n] = ss; ad2[n] = sd;
}

// ---------- 8/9/10) layer-2 edge passes (single head) ----------
__global__ void k_max2(const long long* __restrict__ ei, int E0, int Etot,
                       const float* __restrict__ as, const float* __restrict__ ad,
                       unsigned* __restrict__ mkey) {
    int e = blockIdx.x * blockDim.x + threadIdx.x;
    if (e >= Etot) return;
    int s, d; edge_sd(ei, E0, e, s, d);
    float x  = as[s] + ad[d];
    float lg = x > 0.f ? x : NEG_SLOPE * x;
    atomicMax(&mkey[d], f2key(lg));
}

__global__ void k_exp2(const long long* __restrict__ ei, int E0, int Etot,
                       const float* __restrict__ as, const float* __restrict__ ad,
                       const unsigned* __restrict__ mkey, float* __restrict__ denom,
                       float* __restrict__ ee) {
    int e = blockIdx.x * blockDim.x + threadIdx.x;
    if (e >= Etot) return;
    int s, d; edge_sd(ei, E0, e, s, d);
    float x  = as[s] + ad[d];
    float lg = x > 0.f ? x : NEG_SLOPE * x;
    float ev = __expf(lg - key2f(mkey[d]));
    ee[e] = ev;
    atomicAdd(&denom[d], ev);
}

__global__ void k_agg2(const long long* __restrict__ ei, int E0, int Etot,
                       const float* __restrict__ h2, const float* __restrict__ ee,
                       const float* __restrict__ denom, float* __restrict__ out2) {
    int i = blockIdx.x * blockDim.x + threadIdx.x;   // (e, c)
    if (i >= Etot * 6) return;
    int e = i / 6, c = i - e * 6;
    int s, d; edge_sd(ei, E0, e, s, d);
    float w = ee[e] / denom[d];
    atomicAdd(&out2[(size_t)d * 6 + c], h2[(size_t)s * 6 + c] * w);
}

// ---------- 11) bias + log_softmax over 6 classes ----------
__global__ void k_lsm(const float* __restrict__ out2, const float* __restrict__ b2,
                      float* __restrict__ out, int N) {
    int n = blockIdx.x * blockDim.x + threadIdx.x;
    if (n >= N) return;
    float o[6], m = -3.0e38f;
#pragma unroll
    for (int c = 0; c < 6; ++c) {
        o[c] = out2[(size_t)n * 6 + c] + b2[c];
        m = fmaxf(m, o[c]);
    }
    float s = 0.f;
#pragma unroll
    for (int c = 0; c < 6; ++c) s += __expf(o[c] - m);
    float lse = m + __logf(s);
#pragma unroll
    for (int c = 0; c < 6; ++c) out[(size_t)n * 6 + c] = o[c] - lse;
}

extern "C" void kernel_launch(void* const* d_in, const int* in_sizes, int n_in,
                              void* d_out, int out_size, void* d_ws, size_t ws_size,
                              hipStream_t stream) {
    const float*     x      = (const float*)d_in[0];
    const long long* ei     = (const long long*)d_in[1];   // int64 [2, E0]
    const float*     W1     = (const float*)d_in[2];
    const float*     a1_src = (const float*)d_in[3];
    const float*     a1_dst = (const float*)d_in[4];
    const float*     b1     = (const float*)d_in[5];
    const float*     W2     = (const float*)d_in[6];
    const float*     a2_src = (const float*)d_in[7];
    const float*     a2_dst = (const float*)d_in[8];
    const float*     b2     = (const float*)d_in[9];
    float* out = (float*)d_out;

    const int N    = in_sizes[0] / 128;   // 50000
    const int E0   = in_sizes[1] / 2;     // 800000
    const int Etot = E0 + N;              // with self-loops

    // ---- carve workspace ----
    char* p = (char*)d_ws;
    auto carve = [&](size_t bytes) -> char* {
        char* r = p;
        p += (bytes + 255) & ~(size_t)255;
        return r;
    };
    float*    h1     = (float*)carve((size_t)N * 128 * 4);
    char*     zbeg   = p;                                   // zero-init region start
    float*    out1   = (float*)carve((size_t)N * 128 * 4);
    float*    denom1 = (float*)carve((size_t)N * 4 * 4);
    unsigned* m1k    = (unsigned*)carve((size_t)N * 4 * 4);
    float*    out2   = (float*)carve((size_t)N * 6 * 4);
    float*    denom2 = (float*)carve((size_t)N * 4);
    unsigned* m2k    = (unsigned*)carve((size_t)N * 4);
    size_t    zbytes = (size_t)(p - zbeg);
    float*    a1s    = (float*)carve((size_t)N * 4 * 4);
    float*    a1d    = (float*)carve((size_t)N * 4 * 4);
    float*    ee     = (float*)carve((size_t)Etot * 4 * 4); // reused for layer 2 (Etot)
    float*    h2     = (float*)carve((size_t)N * 6 * 4);
    float*    as2    = (float*)carve((size_t)N * 4);
    float*    ad2    = (float*)carve((size_t)N * 4);
    bf16*     xbf    = (bf16*)carve((size_t)N * 128 * 2);
    bf16*     w1bf   = (bf16*)carve((size_t)128 * 128 * 2);

    const int B = 256;

    // accumulator / key init (graph-capture-safe stream memset)
    hipMemsetAsync(zbeg, 0, zbytes, stream);

    // 0) bf16 conversion
    {
        int nx = N * 128, nw = 128 * 128;
        int n = nx > nw ? nx : nw;
        k_convert<<<CDIV(n, B), B, 0, stream>>>(x, W1, xbf, w1bf, nx, nw);
    }
    // 1) WMMA GEMM: h1 = x @ W1
    {
        int mTiles = CDIV(N, 16);
        int waves  = mTiles * 8;                 // 8 col tiles of 16
        k_gemm1_wmma<<<CDIV(waves, B / 32), B, 0, stream>>>(xbf, w1bf, h1, N);
    }
    // 2) layer-1 attention coefficients
    k_alpha1<<<CDIV(N * 4, B), B, 0, stream>>>(h1, a1_src, a1_dst, a1s, a1d, N);
    // 3-5) layer-1 edge passes
    k_max1<<<CDIV(Etot * 4, B), B, 0, stream>>>(ei, E0, Etot, a1s, a1d, m1k);
    k_exp1<<<CDIV(Etot * 4, B), B, 0, stream>>>(ei, E0, Etot, a1s, a1d, m1k, denom1, ee);
    k_agg1<<<CDIV(Etot * 32, B), B, 0, stream>>>(ei, E0, Etot, h1, ee, denom1, out1);
    // 6) bias + ELU
    k_elu<<<CDIV(N * 128, B), B, 0, stream>>>(out1, b1, N * 128);
    // 7) layer-2 projection + coefficients
    k_proj2<<<CDIV(N, B), B, 0, stream>>>(out1, W2, a2_src, a2_dst, h2, as2, ad2, N);
    // 8-10) layer-2 edge passes (single head)
    k_max2<<<CDIV(Etot, B), B, 0, stream>>>(ei, E0, Etot, as2, ad2, m2k);
    k_exp2<<<CDIV(Etot, B), B, 0, stream>>>(ei, E0, Etot, as2, ad2, m2k, denom2, ee);
    k_agg2<<<CDIV(Etot * 6, B), B, 0, stream>>>(ei, E0, Etot, h2, ee, denom2, out2);
    // 11) bias + log_softmax -> d_out
    k_lsm<<<CDIV(N, B), B, 0, stream>>>(out2, b2, out, N);
}
